// LSTMBinaryClassifier_21028159881631
// MI455X (gfx1250) — compile-verified, matrix operand order checked by reading, not run on hardware
//
#include <hip/hip_runtime.h>

#define BATCH   1024
#define TSEQ    512
#define IN_DIM  24
#define HID     64
#define GATES   256     // 4*HID
#define CLS     2
#define KPAD    32      // IN_DIM padded to WMMA K
#define BTILE   16      // batch rows per workgroup (WMMA M)

typedef __bf16 bf16;
typedef __attribute__((ext_vector_type(16))) __bf16       v16bf;
typedef __attribute__((ext_vector_type(4)))  __bf16       v4bf;
typedef __attribute__((ext_vector_type(8)))  float        v8f;
typedef __attribute__((ext_vector_type(4)))  float        v4f;
typedef __attribute__((ext_vector_type(4)))  unsigned int v4u;

union Frag { v16bf v; v4u q[2]; };

// Load a 16x32 bf16 WMMA fragment (A or B) from a row-major [rows][stride]
// array. Per the CDNA5 ISA layout tables, lane L holds row (L&15), with
// K-chunks [half*8 .. half*8+7] and [16+half*8 .. 23+half*8] contiguously,
// i.e. two 16-byte loads per lane (ds_load_b128 / global_load_b128).
template <typename P>
__device__ __forceinline__ v16bf load_frag(const P* base, int stride, int lane) {
    const int r = lane & 15, half = lane >> 4;
    const P* p = base + r * stride + half * 8;
    Frag u;
    u.q[0] = *(const v4u*)(p);
    u.q[1] = *(const v4u*)(p + 16);
    return u.v;
}

__device__ __forceinline__ v8f wmma_bf16(v16bf a, v16bf b, v8f c) {
    return __builtin_amdgcn_wmma_f32_16x16x32_bf16(
        /*neg_a=*/false, a, /*neg_b=*/false, b,
        /*c_mod=*/(short)0, c, /*reuse_a=*/false, /*reuse_b=*/false);
}

// CDNA5 hardware transcendental tanh (TRANS32 op). The trailing v_nop
// satisfies the ISA's required gap of one op between a TRANS op
// and the first use of its result.
__device__ __forceinline__ float tanh_hw(float x) {
    float r;
    asm("v_tanh_f32 %0, %1\n\tv_nop" : "=v"(r) : "v"(x));
    return r;
}
// sigmoid(x) = 0.5*tanh(0.5x) + 0.5 : mul + tanh + fma, no divide, bounded.
__device__ __forceinline__ float sigm_hw(float x) {
    return __builtin_fmaf(0.5f, tanh_hw(0.5f * x), 0.5f);
}

// ---------------------------------------------------------------------------
// Prep: fp32 weights -> bf16 (K-padded), fold b_ih+b_hh into one bias.
// ---------------------------------------------------------------------------
__global__ void prep_weights(const float* __restrict__ Wih0, const float* __restrict__ Whh0,
                             const float* __restrict__ bih0, const float* __restrict__ bhh0,
                             const float* __restrict__ Wih1, const float* __restrict__ Whh1,
                             const float* __restrict__ bih1, const float* __restrict__ bhh1,
                             bf16* __restrict__ wih0b, bf16* __restrict__ whh0b,
                             bf16* __restrict__ wih1b, bf16* __restrict__ whh1b,
                             float* __restrict__ bias0, float* __restrict__ bias1) {
    const int n = threadIdx.x;  // 0..255 : one gate row per thread
    for (int k = 0; k < KPAD; ++k)
        wih0b[n * KPAD + k] = (k < IN_DIM) ? (bf16)Wih0[n * IN_DIM + k] : (bf16)0.0f;
    for (int k = 0; k < HID; ++k) {
        whh0b[n * HID + k] = (bf16)Whh0[n * HID + k];
        wih1b[n * HID + k] = (bf16)Wih1[n * HID + k];
        whh1b[n * HID + k] = (bf16)Whh1[n * HID + k];
    }
    bias0[n] = bih0[n] + bhh0[n];
    bias1[n] = bih1[n] + bhh1[n];
}

// ---------------------------------------------------------------------------
// Fused 2-layer LSTM scan + final FC. One workgroup = 16 sequences.
// ---------------------------------------------------------------------------
__global__ __launch_bounds__(256) void lstm_fused(
    const float* __restrict__ x,
    const bf16* __restrict__ wih0b, const bf16* __restrict__ whh0b,
    const bf16* __restrict__ wih1b, const bf16* __restrict__ whh1b,
    const float* __restrict__ bias0, const float* __restrict__ bias1,
    const float* __restrict__ Wfc, const float* __restrict__ bfc,
    float* __restrict__ out)
{
    __shared__ float c1[BTILE][HID];
    __shared__ float c2[BTILE][HID];
    __shared__ float h2[BTILE][HID];
    __shared__ bf16  h1b[BTILE][HID];
    __shared__ bf16  h2b[BTILE][HID];
    __shared__ bf16  xb[BTILE][KPAD];
    __shared__ float gates[BTILE][GATES];

    const int tid  = threadIdx.x;
    const int lane = tid & 31;
    const int wave = tid >> 5;                 // 0..7
    const int b0   = blockIdx.x * BTILE;

    // ---- zero state ----
    for (int i = tid; i < BTILE * HID; i += 256) {
        ((float*)c1)[i] = 0.f; ((float*)c2)[i] = 0.f; ((float*)h2)[i] = 0.f;
        ((bf16*)h1b)[i] = (bf16)0.f; ((bf16*)h2b)[i] = (bf16)0.f;
    }
    for (int i = tid; i < BTILE * KPAD; i += 256) ((bf16*)xb)[i] = (bf16)0.f;

    // ---- per-wave persistent weight B-fragments (2 gate-column tiles/wave) ----
    const int n0 = (wave * 2) * 16;
    const int n1 = (wave * 2 + 1) * 16;
    const v16bf wih0_a  = load_frag(wih0b + n0 * KPAD, KPAD, lane);
    const v16bf wih0_b  = load_frag(wih0b + n1 * KPAD, KPAD, lane);
    const v16bf whh0_a0 = load_frag(whh0b + n0 * HID,       HID, lane);
    const v16bf whh0_a1 = load_frag(whh0b + n0 * HID + 32,  HID, lane);
    const v16bf whh0_b0 = load_frag(whh0b + n1 * HID,       HID, lane);
    const v16bf whh0_b1 = load_frag(whh0b + n1 * HID + 32,  HID, lane);
    const v16bf wih1_a0 = load_frag(wih1b + n0 * HID,       HID, lane);
    const v16bf wih1_a1 = load_frag(wih1b + n0 * HID + 32,  HID, lane);
    const v16bf wih1_b0 = load_frag(wih1b + n1 * HID,       HID, lane);
    const v16bf wih1_b1 = load_frag(wih1b + n1 * HID + 32,  HID, lane);
    const v16bf whh1_a0 = load_frag(whh1b + n0 * HID,       HID, lane);
    const v16bf whh1_a1 = load_frag(whh1b + n0 * HID + 32,  HID, lane);
    const v16bf whh1_b0 = load_frag(whh1b + n1 * HID,       HID, lane);
    const v16bf whh1_b1 = load_frag(whh1b + n1 * HID + 32,  HID, lane);

    const int   ncol   = lane & 15;
    const float biasA0 = bias0[n0 + ncol];
    const float biasB0 = bias0[n1 + ncol];
    const float biasA1 = bias1[n0 + ncol];
    const float biasB1 = bias1[n1 + ncol];
    const int   mbase  = (lane >> 4) * 8;

    // elementwise tiling: each thread owns row em, 4 consecutive cols ej..ej+3
    const int em = tid >> 4;
    const int ej = (tid & 15) * 4;

    // ---- stage x_0 ----
    for (int i = tid; i < BTILE * IN_DIM; i += 256) {
        const int m = i / IN_DIM, k = i - m * IN_DIM;
        xb[m][k] = (bf16)x[(size_t)(b0 + m) * TSEQ * IN_DIM + k];
    }
    __syncthreads();

    for (int t = 0; t < TSEQ; ++t) {
        // ---- layer 0: gates = bias + x_t @ Wih0^T + h1 @ Whh0^T ----
        {
            const v16bf ax  = load_frag(&xb[0][0],   KPAD, lane);
            const v16bf ah0 = load_frag(&h1b[0][0],  HID,  lane);
            const v16bf ah1 = load_frag(&h1b[0][32], HID,  lane);
            v8f accA, accB;
            for (int r = 0; r < 8; ++r) { accA[r] = biasA0; accB[r] = biasB0; }
            accA = wmma_bf16(ax,  wih0_a,  accA);
            accA = wmma_bf16(ah0, whh0_a0, accA);
            accA = wmma_bf16(ah1, whh0_a1, accA);
            accB = wmma_bf16(ax,  wih0_b,  accB);
            accB = wmma_bf16(ah0, whh0_b0, accB);
            accB = wmma_bf16(ah1, whh0_b1, accB);
            for (int r = 0; r < 8; ++r) {
                gates[mbase + r][n0 + ncol] = accA[r];
                gates[mbase + r][n1 + ncol] = accB[r];
            }
        }
        __syncthreads();

        // ---- layer 0 cell update (4 cells/thread, vectorized LDS) ----
        {
            v4f gi = *(const v4f*)&gates[em][ej];
            v4f gf = *(const v4f*)&gates[em][HID + ej];
            v4f gg = *(const v4f*)&gates[em][2 * HID + ej];
            v4f go = *(const v4f*)&gates[em][3 * HID + ej];
            v4f cc = *(const v4f*)&c1[em][ej];
            v4bf hb;
            for (int q = 0; q < 4; ++q) {
                const float ig = sigm_hw(gi[q]);
                const float fg = sigm_hw(gf[q]);
                const float g2 = tanh_hw(gg[q]);
                const float og = sigm_hw(go[q]);
                const float c  = fg * cc[q] + ig * g2;
                cc[q] = c;
                hb[q] = (bf16)(og * tanh_hw(c));
            }
            *(v4f*)&c1[em][ej]  = cc;
            *(v4bf*)&h1b[em][ej] = hb;
        }
        __syncthreads();

        // ---- layer 1: gates = bias + h1_t @ Wih1^T + h2 @ Whh1^T ----
        // (also stage x_{t+1}: xb is not read in this phase)
        {
            const v16bf a10 = load_frag(&h1b[0][0],  HID, lane);
            const v16bf a11 = load_frag(&h1b[0][32], HID, lane);
            const v16bf a20 = load_frag(&h2b[0][0],  HID, lane);
            const v16bf a21 = load_frag(&h2b[0][32], HID, lane);
            v8f accA, accB;
            for (int r = 0; r < 8; ++r) { accA[r] = biasA1; accB[r] = biasB1; }
            accA = wmma_bf16(a10, wih1_a0, accA);
            accA = wmma_bf16(a11, wih1_a1, accA);
            accA = wmma_bf16(a20, whh1_a0, accA);
            accA = wmma_bf16(a21, whh1_a1, accA);
            accB = wmma_bf16(a10, wih1_b0, accB);
            accB = wmma_bf16(a11, wih1_b1, accB);
            accB = wmma_bf16(a20, whh1_b0, accB);
            accB = wmma_bf16(a21, whh1_b1, accB);
            for (int r = 0; r < 8; ++r) {
                gates[mbase + r][n0 + ncol] = accA[r];
                gates[mbase + r][n1 + ncol] = accB[r];
            }

            if (t + 1 < TSEQ) {
                for (int i = tid; i < BTILE * IN_DIM; i += 256) {
                    const int m = i / IN_DIM, k = i - m * IN_DIM;
                    xb[m][k] = (bf16)x[((size_t)(b0 + m) * TSEQ + (t + 1)) * IN_DIM + k];
                }
                if (t + 2 < TSEQ && tid < BTILE)  // pull t+2 toward L2/WGP$
                    __builtin_prefetch(&x[((size_t)(b0 + tid) * TSEQ + (t + 2)) * IN_DIM], 0, 3);
            }
        }
        __syncthreads();

        // ---- layer 1 cell update ----
        {
            v4f gi = *(const v4f*)&gates[em][ej];
            v4f gf = *(const v4f*)&gates[em][HID + ej];
            v4f gg = *(const v4f*)&gates[em][2 * HID + ej];
            v4f go = *(const v4f*)&gates[em][3 * HID + ej];
            v4f cc = *(const v4f*)&c2[em][ej];
            v4bf hb;
            v4f  hf;
            for (int q = 0; q < 4; ++q) {
                const float ig = sigm_hw(gi[q]);
                const float fg = sigm_hw(gf[q]);
                const float g2 = tanh_hw(gg[q]);
                const float og = sigm_hw(go[q]);
                const float c  = fg * cc[q] + ig * g2;
                cc[q] = c;
                const float h  = og * tanh_hw(c);
                hf[q] = h;
                hb[q] = (bf16)h;
            }
            *(v4f*)&c2[em][ej]   = cc;
            *(v4f*)&h2[em][ej]   = hf;
            *(v4bf*)&h2b[em][ej] = hb;
        }
        __syncthreads();
    }

    // ---- final FC: out = h2_last @ Wfc^T + bfc  (16 x 2) ----
    if (tid < BTILE * CLS) {
        const int m = tid >> 1, c = tid & 1;
        float s = bfc[c];
        for (int k = 0; k < HID; ++k) s += h2[m][k] * Wfc[c * HID + k];
        out[(size_t)(b0 + m) * CLS + c] = s;
    }
}

// ---------------------------------------------------------------------------
extern "C" void kernel_launch(void* const* d_in, const int* in_sizes, int n_in,
                              void* d_out, int out_size, void* d_ws, size_t ws_size,
                              hipStream_t stream) {
    (void)in_sizes; (void)n_in; (void)out_size; (void)ws_size;

    const float* x    = (const float*)d_in[0];
    const float* Wih0 = (const float*)d_in[1];
    const float* Whh0 = (const float*)d_in[2];
    const float* bih0 = (const float*)d_in[3];
    const float* bhh0 = (const float*)d_in[4];
    const float* Wih1 = (const float*)d_in[5];
    const float* Whh1 = (const float*)d_in[6];
    const float* bih1 = (const float*)d_in[7];
    const float* bhh1 = (const float*)d_in[8];
    const float* Wfc  = (const float*)d_in[9];
    const float* bfc  = (const float*)d_in[10];

    char* ws = (char*)d_ws;
    bf16*  wih0b = (bf16*)(ws);              // 256*32*2  = 16384
    bf16*  whh0b = (bf16*)(ws + 16384);      // 256*64*2  = 32768
    bf16*  wih1b = (bf16*)(ws + 49152);      // 32768
    bf16*  whh1b = (bf16*)(ws + 81920);      // 32768
    float* bias0 = (float*)(ws + 114688);    // 1024
    float* bias1 = (float*)(ws + 115712);    // 1024

    prep_weights<<<1, 256, 0, stream>>>(Wih0, Whh0, bih0, bhh0,
                                        Wih1, Whh1, bih1, bhh1,
                                        wih0b, whh0b, wih1b, whh1b,
                                        bias0, bias1);

    lstm_fused<<<BATCH / BTILE, 256, 0, stream>>>(
        x, wih0b, whh0b, wih1b, whh1b, bias0, bias1, Wfc, bfc, (float*)d_out);
}
